// Generator_21981642621372
// MI455X (gfx1250) — matrix operand order repaired and checked by reading, not compiled, not run on hardware
//
#include <hip/hip_runtime.h>
#include <hip/hip_bf16.h>

typedef __attribute__((ext_vector_type(16))) _Float16 v16h;
typedef __attribute__((ext_vector_type(8)))  float    v8f;

#define DEV __device__ __forceinline__

DEV float lrelu(float x) { return x >= 0.f ? x : 0.2f * x; }

// A-fragment (f16 16x32) element (lane, i) -> k offset inside the 32-wide K tile.
// Contiguous order: lanes 0-15 hold K0..15, lanes 16-31 hold K16..31.
DEV int a_frag_k(int lane, int i) {
    int half = lane >> 4;
    return i + half * 16;
}

// ---------------------------------------------------------------------------
// K1: init conv (1,512,4,4)->(256,4,4), lrelu, bilinear x2 -> u1[256,8,8] f32.
// One block per output channel o.
// ---------------------------------------------------------------------------
__global__ void k_init(const float* __restrict__ init_const,
                       const float* __restrict__ icw,
                       const float* __restrict__ icb,
                       float* __restrict__ u1)
{
    __shared__ float red[64];
    __shared__ float base[16];
    int o = blockIdx.x;        // 0..255
    int t = threadIdx.x;       // 0..63
    int q = t >> 4, p = t & 15;
    int py = p >> 2, px = p & 3;
    float acc = 0.f;
    for (int ic = q * 128; ic < q * 128 + 128; ++ic) {
        const float* wv = icw + (size_t)(o * 512 + ic) * 9;
        const float* xv = init_const + ic * 16;
        #pragma unroll
        for (int dy = 0; dy < 3; ++dy) {
            int yy = py + dy - 1;
            if (yy < 0 || yy > 3) continue;
            #pragma unroll
            for (int dx = 0; dx < 3; ++dx) {
                int xx = px + dx - 1;
                if (xx < 0 || xx > 3) continue;
                acc += wv[dy * 3 + dx] * xv[yy * 4 + xx];
            }
        }
    }
    red[t] = acc;
    __syncthreads();
    if (t < 16) base[t] = lrelu(red[t] + red[t + 16] + red[t + 32] + red[t + 48] + icb[o]);
    __syncthreads();
    // bilinear 4 -> 8, half-pixel centers: src = i*0.5 - 0.25
    int y = t >> 3, x = t & 7;
    int ty = (y - 1) >> 1, tx = (x - 1) >> 1;
    int y0 = ty < 0 ? 0 : ty, y1 = ty + 1 > 3 ? 3 : ty + 1;
    int x0 = tx < 0 ? 0 : tx, x1 = tx + 1 > 3 ? 3 : tx + 1;
    float fy = (y & 1) ? 0.25f : 0.75f;
    float fx = (x & 1) ? 0.25f : 0.75f;
    float v = (1.f - fy) * ((1.f - fx) * base[y0 * 4 + x0] + fx * base[y0 * 4 + x1])
            +        fy  * ((1.f - fx) * base[y1 * 4 + x0] + fx * base[y1 * 4 + x1]);
    u1[o * 64 + t] = v;
}

// ---------------------------------------------------------------------------
// K2: M1[c,(o,p)] = sum_k W1[o,c,k]*u1pad[c,p+k], packed straight into WMMA
// B-fragment layout: m1f[((tileN*8 + kTile)*32 + lane)*16 + i], f16.
// Grid: 512 tileN * 8 kTile blocks, 32 threads.
// ---------------------------------------------------------------------------
__global__ void k_m1frag(const float* __restrict__ u1,
                         const float* __restrict__ c1w,
                         _Float16* __restrict__ m1f)
{
    int bx = blockIdx.x;
    int tileN = bx >> 3, kTile = bx & 7;
    int lane = threadIdx.x;
    int nloc = lane & 15;
    int kBase = kTile * 32 + (lane >> 4) * 16;
    int nG = tileN * 16 + nloc;         // n = o*64 + p
    int o = nG >> 6, p = nG & 63;
    int py = p >> 3, px = p & 7;
    _Float16* dst = m1f + ((size_t)(tileN * 8 + kTile) * 32 + lane) * 16;
    #pragma unroll
    for (int i = 0; i < 16; ++i) {
        int c = kBase + i;              // K == input channel (256)
        const float* wv = c1w + (size_t)(o * 256 + c) * 9;
        float acc = 0.f;
        #pragma unroll
        for (int dy = 0; dy < 3; ++dy) {
            int yy = py + dy - 1;
            if ((unsigned)yy > 7u) continue;
            #pragma unroll
            for (int dx = 0; dx < 3; ++dx) {
                int xx = px + dx - 1;
                if ((unsigned)xx > 7u) continue;
                acc += wv[dy * 3 + dx] * u1[c * 64 + yy * 8 + xx];
            }
        }
        dst[i] = (_Float16)acc;
    }
}

// ---------------------------------------------------------------------------
// K3: style GEMVs: style1[b,256] and style2[b,128], stored f16.
// ---------------------------------------------------------------------------
__global__ void k_styles(const float* __restrict__ w,
                         const float* __restrict__ a1w, const float* __restrict__ a1b,
                         const float* __restrict__ a2w, const float* __restrict__ a2b,
                         _Float16* __restrict__ s1h, _Float16* __restrict__ s2h)
{
    int idx = blockIdx.x * blockDim.x + threadIdx.x;
    if (idx >= 2048 * 384) return;
    int b = idx / 384, j = idx % 384;
    const float* wb = w + (size_t)b * 512;
    if (j < 256) {
        const float* av = a1w + (size_t)j * 512;
        float acc = a1b[j];
        for (int k = 0; k < 512; ++k) acc += wb[k] * av[k];
        s1h[b * 256 + j] = (_Float16)acc;
    } else {
        int j2 = j - 256;
        const float* av = a2w + (size_t)j2 * 512;
        float acc = a2b[j2];
        for (int k = 0; k < 512; ++k) acc += wb[k] * av[k];
        s2h[b * 128 + j2] = (_Float16)acc;
    }
}

// K3b: w2[o,c] = sum_k W[o,c,k]^2 for both conv layers.
__global__ void k_w2(const float* __restrict__ c1w, const float* __restrict__ c2w,
                     float* __restrict__ w21, float* __restrict__ w22)
{
    int idx = blockIdx.x * blockDim.x + threadIdx.x;
    if (idx < 128 * 256) {
        const float* p = c1w + (size_t)idx * 9;
        float a = 0.f;
        #pragma unroll
        for (int i = 0; i < 9; ++i) a += p[i] * p[i];
        w21[idx] = a;
    } else if (idx < 128 * 256 + 64 * 128) {
        int i2 = idx - 128 * 256;
        const float* p = c2w + (size_t)i2 * 9;
        float a = 0.f;
        #pragma unroll
        for (int i = 0; i < 9; ++i) a += p[i] * p[i];
        w22[i2] = a;
    }
}

// K4: demod[b,o] = rsqrt(sum_c style^2 * w2[o,c] + 1e-8)
__global__ void k_demod(const _Float16* __restrict__ s1h, const _Float16* __restrict__ s2h,
                        const float* __restrict__ w21, const float* __restrict__ w22,
                        float* __restrict__ dm1, float* __restrict__ dm2)
{
    int idx = blockIdx.x * blockDim.x + threadIdx.x;
    if (idx < 2048 * 128) {
        int b = idx >> 7, o = idx & 127;
        float acc = 0.f;
        for (int c = 0; c < 256; ++c) { float s = (float)s1h[b * 256 + c]; acc += s * s * w21[o * 256 + c]; }
        dm1[idx] = rsqrtf(acc + 1e-8f);
    } else if (idx < 2048 * 128 + 2048 * 64) {
        int i2 = idx - 2048 * 128;
        int b = i2 >> 6, o = i2 & 63;
        float acc = 0.f;
        for (int c = 0; c < 128; ++c) { float s = (float)s2h[b * 128 + c]; acc += s * s * w22[o * 128 + c]; }
        dm2[i2] = rsqrtf(acc + 1e-8f);
    }
}

// ---------------------------------------------------------------------------
// K5: layer-1 collapsed GEMM: x1[b,n] = lrelu(style1[b,:] @ M1[:,n] * dm1 + b1)
// M=2048 (batch), K=256, N=8192. One wave per 16x64 tile.
// ---------------------------------------------------------------------------
__global__ void __launch_bounds__(32) k_layer1(
    const _Float16* __restrict__ s1h,
    const _Float16* __restrict__ m1f,
    const float* __restrict__ dm1,
    const float* __restrict__ c1b,
    _Float16* __restrict__ x1h)
{
    int b0 = blockIdx.x * 16;
    int oc = blockIdx.y;                // output channel 0..127 (== n/64 group)
    int lane = threadIdx.x;
    int half = lane >> 4, col = lane & 15;
    v8f zero;
    #pragma unroll
    for (int r = 0; r < 8; ++r) zero[r] = 0.f;
    v8f acc[4] = { zero, zero, zero, zero };
    const v16h* Bf = (const v16h*)m1f;
    for (int kTile = 0; kTile < 8; ++kTile) {
        v16h a;
        #pragma unroll
        for (int i = 0; i < 16; ++i) {
            int k = kTile * 32 + a_frag_k(lane, i);
            a[i] = s1h[(size_t)(b0 + col) * 256 + k];
        }
        #pragma unroll
        for (int j = 0; j < 4; ++j) {
            int tileN = oc * 4 + j;
            v16h bf = Bf[(size_t)(tileN * 8 + kTile) * 32 + lane];
            acc[j] = __builtin_amdgcn_wmma_f32_16x16x32_f16(
                false, a, false, bf, (short)0, acc[j], false, false);
        }
    }
    #pragma unroll
    for (int j = 0; j < 4; ++j) {
        #pragma unroll
        for (int r = 0; r < 8; ++r) {
            int bb = b0 + r + 8 * half;
            int pp = j * 16 + col;
            float v = acc[j][r] * dm1[bb * 128 + oc] + c1b[oc];
            x1h[(size_t)bb * 8192 + oc * 64 + pp] = (_Float16)lrelu(v);
        }
    }
}

// ---------------------------------------------------------------------------
// K6: pack conv2 weights into A-fragment layout. Layer-2 K order is TAP-MAJOR:
// k = tap*128 + c  (1152 = 9 taps * 128 ch = 36 kTiles of 32; each kTile has a
// single uniform tap since 32 | 128). No divisions needed anywhere.
// ---------------------------------------------------------------------------
__global__ void k_a2frag(const float* __restrict__ c2w, _Float16* __restrict__ a2f)
{
    int bx = blockIdx.x;                // 4 mTile * 36 kTile
    int mTile = bx / 36, kTile = bx % 36;
    int lane = threadIdx.x;
    int m = lane & 15;
    _Float16* dst = a2f + ((size_t)(mTile * 36 + kTile) * 32 + lane) * 16;
    #pragma unroll
    for (int i = 0; i < 16; ++i) {
        int k = kTile * 32 + a_frag_k(lane, i);
        int c = k & 127, tap = k >> 7;
        int o = mTile * 16 + m;
        dst[i] = (_Float16)c2w[(size_t)(o * 128 + c) * 9 + tap];
    }
}

// ---------------------------------------------------------------------------
// K7: layer 2, one workgroup per batch element. Stage upsampled+modulated
// activation CHANNEL-LAST ([18*18 pos][128 ch], zero halo) as f16 in LDS so a
// B fragment is one contiguous 32-byte LDS read per lane. Implicit-GEMM conv:
// M=64 out ch (4 waves x 16), N=256 spatial (16 row tiles, y-blocked by 8),
// K=1152 tap-major. Inner body: 1 global A load + 8x{2 ds_load_b128 + WMMA}.
// ---------------------------------------------------------------------------
__global__ void __launch_bounds__(128) k_layer2(
    const _Float16* __restrict__ x1h,
    const _Float16* __restrict__ s2h,
    const _Float16* __restrict__ a2f,
    const float* __restrict__ dm2,
    const float* __restrict__ c2b,
    float* __restrict__ out)
{
    __shared__ _Float16 sraw[128 * 64];                      // 16 KB
    __shared__ __align__(32) _Float16 sxm[18 * 18 * 128];    // 81 KB, channel-last
    int b = blockIdx.x, tid = threadIdx.x;

    const _Float16* src = x1h + (size_t)b * 8192;
    for (int i = tid; i < 8192; i += 128) sraw[i] = src[i];
    __syncthreads();

    {   // upsample x2 (half-pixel) + modulate + zero pad; one channel/thread,
        // writes are consecutive across lanes (conflict-free).
        int c = tid;
        float s = (float)s2h[b * 128 + c];
        const _Float16* r = sraw + c * 64;
        for (int y = 0; y < 18; ++y) {
            for (int x = 0; x < 18; ++x) {
                float v = 0.f;
                if (y > 0 && y < 17 && x > 0 && x < 17) {
                    int oy = y - 1, ox = x - 1;
                    int ty = (oy - 1) >> 1, tx = (ox - 1) >> 1;
                    int y0 = ty < 0 ? 0 : ty, y1 = ty + 1 > 7 ? 7 : ty + 1;
                    int x0 = tx < 0 ? 0 : tx, x1 = tx + 1 > 7 ? 7 : tx + 1;
                    float fy = (oy & 1) ? 0.25f : 0.75f;
                    float fx = (ox & 1) ? 0.25f : 0.75f;
                    float v00 = (float)r[y0 * 8 + x0], v01 = (float)r[y0 * 8 + x1];
                    float v10 = (float)r[y1 * 8 + x0], v11 = (float)r[y1 * 8 + x1];
                    v = ((1.f - fy) * ((1.f - fx) * v00 + fx * v01)
                       +        fy  * ((1.f - fx) * v10 + fx * v11)) * s;
                }
                sxm[(y * 18 + x) * 128 + c] = (_Float16)v;
            }
        }
    }
    __syncthreads();

    int mTile = tid >> 5;       // wave id = output-channel tile
    int lane = tid & 31;
    int half = lane >> 4, col = lane & 15;
    const v16h* Af = (const v16h*)a2f;
    for (int yc = 0; yc < 16; yc += 8) {
        v8f acc[8];
        #pragma unroll
        for (int yi = 0; yi < 8; ++yi)
            #pragma unroll
            for (int r8 = 0; r8 < 8; ++r8) acc[yi][r8] = 0.f;

        for (int kTile = 0; kTile < 36; ++kTile) {
            v16h a = Af[(size_t)(mTile * 36 + kTile) * 32 + lane];
            int tap = kTile >> 2;                 // uniform within kTile
            int dy = tap / 3, dx = tap - dy * 3;  // tap is loop-derived, cheap
            int kLocal = (kTile & 3) * 32 + half * 16;
            #pragma unroll
            for (int yi = 0; yi < 8; ++yi) {
                int y = yc + yi;
                const v16h* bp = (const v16h*)&sxm[((y + dy) * 18 + (col + dx)) * 128 + kLocal];
                acc[yi] = __builtin_amdgcn_wmma_f32_16x16x32_f16(
                    false, a, false, *bp, (short)0, acc[yi], false, false);
            }
        }
        #pragma unroll
        for (int yi = 0; yi < 8; ++yi) {
            #pragma unroll
            for (int r8 = 0; r8 < 8; ++r8) {
                int o = mTile * 16 + r8 + 8 * half;
                float v = acc[yi][r8] * dm2[b * 64 + o] + c2b[o];
                out[(size_t)b * 16384 + o * 256 + (yc + yi) * 16 + col] = lrelu(v);
            }
        }
    }
}

// ---------------------------------------------------------------------------
extern "C" void kernel_launch(void* const* d_in, const int* in_sizes, int n_in,
                              void* d_out, int out_size, void* d_ws, size_t ws_size,
                              hipStream_t stream)
{
    const float* w   = (const float*)d_in[0];
    const float* ic  = (const float*)d_in[1];
    const float* icw = (const float*)d_in[2];
    const float* icb = (const float*)d_in[3];
    const float* a1w = (const float*)d_in[4];
    const float* a1b = (const float*)d_in[5];
    const float* c1w = (const float*)d_in[6];
    const float* c1b = (const float*)d_in[7];
    const float* a2w = (const float*)d_in[8];
    const float* a2b = (const float*)d_in[9];
    const float* c2w = (const float*)d_in[10];
    const float* c2b = (const float*)d_in[11];

    char* ws = (char*)d_ws;
    size_t off = 0;
    auto alloc = [&](size_t bytes) -> void* {
        void* p = ws + off;
        off += (bytes + 255) & ~(size_t)255;
        return p;
    };
    float*    u1  = (float*)   alloc(256 * 64 * 4);
    _Float16* m1f = (_Float16*)alloc((size_t)2097152 * 2);   // 512 tileN * 8 kTile * 512 halfs
    _Float16* s1h = (_Float16*)alloc((size_t)2048 * 256 * 2);
    _Float16* s2h = (_Float16*)alloc((size_t)2048 * 128 * 2);
    float*    w21 = (float*)   alloc(128 * 256 * 4);
    float*    w22 = (float*)   alloc(64 * 128 * 4);
    float*    dm1 = (float*)   alloc((size_t)2048 * 128 * 4);
    float*    dm2 = (float*)   alloc((size_t)2048 * 64 * 4);
    _Float16* x1h = (_Float16*)alloc((size_t)2048 * 8192 * 2);
    _Float16* a2f = (_Float16*)alloc((size_t)73728 * 2);
    (void)ws_size; (void)in_sizes; (void)n_in; (void)out_size;

    float* outp = (float*)d_out;

    k_init  <<<256, 64, 0, stream>>>(ic, icw, icb, u1);
    k_m1frag<<<4096, 32, 0, stream>>>(u1, c1w, m1f);
    k_styles<<<3072, 256, 0, stream>>>(w, a1w, a1b, a2w, a2b, s1h, s2h);
    k_w2    <<<160, 256, 0, stream>>>(c1w, c2w, w21, w22);
    k_demod <<<1536, 256, 0, stream>>>(s1h, s2h, w21, w22, dm1, dm2);
    k_layer1<<<dim3(128, 128), 32, 0, stream>>>(s1h, m1f, dm1, c1b, x1h);
    k_a2frag<<<144, 32, 0, stream>>>(c2w, a2f);
    k_layer2<<<2048, 128, 0, stream>>>(x1h, s2h, a2f, dm2, c2b, outp);
}
